// OmniBrainV8_83940840833560
// MI455X (gfx1250) — matrix operand order repaired and checked by reading, not compiled, not run on hardware
//
#include <hip/hip_runtime.h>
#include <hip/hip_bf16.h>
#include <cstdint>

// ---------------------------------------------------------------------------
// CDNA5 (gfx1250): all matmul-shaped work via v_wmma_f32_16x16x32_bf16.
// Square GEMMs stage tiles LDS-side with the Tensor Data Mover
// (tensor_load_to_lds + s_wait_tensorcnt). Convs are implicit GEMM with
// branchless clamped-address im2col gathers. eigvalsh = parallel Jacobi on a
// 256x256 matrix resident in LDS (CDNA5's 320KB WGP LDS).
// ---------------------------------------------------------------------------

typedef __attribute__((ext_vector_type(16))) __bf16       v16bf;
typedef __attribute__((ext_vector_type(8)))  float        v8f;
typedef __attribute__((ext_vector_type(4)))  unsigned int v4u;
typedef __attribute__((ext_vector_type(8)))  int          v8i;
typedef __attribute__((ext_vector_type(4)))  int          v4i;

#if defined(__HIP_DEVICE_COMPILE__) && __has_builtin(__builtin_amdgcn_tensor_load_to_lds)
#define HAVE_TDM 1
#else
#define HAVE_TDM 0
#endif

__device__ __forceinline__ v8f wmma_bf16(v16bf a, v16bf b, v8f c) {
  // (neg_a, A, neg_b, B, c_mod, C, reuse_a, reuse_b)
  return __builtin_amdgcn_wmma_f32_16x16x32_bf16(false, a, false, b, (short)0, c, false, false);
}

#if HAVE_TDM
// Build a 2D Tensor-DMA descriptor (D#) per CDNA5 ISA
// and issue TENSOR_LOAD_TO_LDS. All inputs are wave-uniform.
//   group0: [1:0]=count=1, [63:32]=lds_addr, [120:64]=global_addr, [127:126]=type=2
//   group1: data_size(4B)=2<<16, tensor_dim0/1, tile_dim0/1, tensor_dim0_stride
__device__ __forceinline__ void tdm_load_2d(const float* gptr, unsigned lds_off,
                                            unsigned tensor_d0, unsigned tensor_d1,
                                            unsigned long long stride0,
                                            unsigned tile_d0, unsigned tile_d1)
{
  unsigned long long ga = (unsigned long long)(uintptr_t)gptr;
  v4u g0;
  g0[0] = 1u;                                             // count=1 (valid), user mode
  g0[1] = lds_off;                                        // LDS byte address
  g0[2] = (unsigned)(ga & 0xffffffffu);                   // global_addr[31:0]
  g0[3] = (unsigned)((ga >> 32) & 0x01ffffffu) | (2u << 30); // addr[56:32] | type=2
  v8i g1;
  g1[0] = (int)(2u << 16);                                // workgroup_mask=0, data_size=4B
  g1[1] = (int)((tensor_d0 & 0xffffu) << 16);             // tensor_dim0[15:0] @ bits 63:48
  g1[2] = (int)(((tensor_d0 >> 16) & 0xffffu) | ((tensor_d1 & 0xffffu) << 16));
  g1[3] = (int)(((tensor_d1 >> 16) & 0xffffu) | (tile_d0 << 16));  // tile_dim0 @127:112
  g1[4] = (int)(tile_d1 & 0xffffu);                       // tile_dim1 @143:128
  g1[5] = (int)(stride0 & 0xffffffffu);                   // tensor_dim0_stride lo
  g1[6] = (int)((stride0 >> 32) & 0xffffu);               // tensor_dim0_stride hi
  g1[7] = 0;
  v4i gz = {0, 0, 0, 0};
#if __clang_major__ >= 23
  v8i gz8 = {0, 0, 0, 0, 0, 0, 0, 0};
  __builtin_amdgcn_tensor_load_to_lds(g0, g1, gz, gz, gz8, 0);
#else
  __builtin_amdgcn_tensor_load_to_lds(g0, g1, gz, gz, 0);
#endif
}
#endif

// ---------------------------------------------------------------------------
// TDM-staged GEMM: out[m,n] = sum_k A[m,k]*W[n,k] + bias[n], N % 64 == 0.
// Block = 4 waves; per block tile = 16(M) x 64(N). Per K-step wave0 DMAs the
// A tile (16x32 f32) and W tile (64x32 f32) into LDS, waits TENSORcnt, then
// all waves build fragments from LDS (ds_load) and issue WMMA.
// ---------------------------------------------------------------------------
__global__ void gemm_xwT_tdm_kernel(const float* __restrict__ A, int lda,
                                    const float* __restrict__ Wt, int ldw,
                                    const float* __restrict__ bias,
                                    float* __restrict__ out, int ldo,
                                    int M, int N, int K)
{
  __shared__ float As[16 * 32];
  __shared__ float Ws[64 * 32];
  int wave = threadIdx.x >> 5;
  int lane = threadIdx.x & 31;
  int nbl  = N >> 6;
  int mt   = blockIdx.x / nbl;
  int nb   = blockIdx.x % nbl;
  int m0   = mt << 4;
  int n0b  = nb << 6;
  int l = lane & 15, h = lane >> 4;

  v8f acc = {};
  for (int k0 = 0; k0 < K; k0 += 32) {
#if HAVE_TDM
    if (wave == 0) {
      tdm_load_2d(A + (size_t)m0 * lda + k0, (unsigned)(uintptr_t)&As[0],
                  (unsigned)K, (unsigned)M, (unsigned long long)lda, 32u, 16u);
      tdm_load_2d(Wt + (size_t)n0b * ldw + k0, (unsigned)(uintptr_t)&Ws[0],
                  (unsigned)K, (unsigned)N, (unsigned long long)ldw, 32u, 64u);
      __builtin_amdgcn_s_wait_tensorcnt(0);
    }
#else
    for (int i = threadIdx.x; i < 16 * 32; i += 128) {
      int rr = i >> 5, kk = i & 31;
      As[i] = A[(size_t)(m0 + rr) * lda + k0 + kk];
    }
    for (int i = threadIdx.x; i < 64 * 32; i += 128) {
      int rr = i >> 5, kk = i & 31;
      Ws[i] = Wt[(size_t)(n0b + rr) * ldw + k0 + kk];
    }
#endif
    __syncthreads();
    v16bf af, bf;
#pragma unroll
    for (int j = 0; j < 8; ++j) {
      af[j]     = (__bf16)As[l * 32 + h * 8 + j];
      af[8 + j] = (__bf16)As[l * 32 + 16 + h * 8 + j];
    }
    int wr = (wave << 4) + l;
#pragma unroll
    for (int j = 0; j < 16; ++j)
      bf[j] = (__bf16)Ws[wr * 32 + (j < 8 ? h * 8 + j : 16 + h * 8 + (j - 8))];
    acc = wmma_bf16(af, bf, acc);
    __syncthreads();
  }

  int nc = n0b + (wave << 4) + l;
  float bv = bias[nc];
#pragma unroll
  for (int r = 0; r < 8; ++r)
    out[(size_t)(m0 + r + 8 * h) * ldo + nc] = acc[r] + bv;
}

// ---------------------------------------------------------------------------
// Direct GEMM (ragged N, classifier): branchless Nvalid guard.
// ---------------------------------------------------------------------------
__global__ void gemm_xwT_kernel(const float* __restrict__ A, int lda,
                                const float* __restrict__ Wt, int ldw,
                                const float* __restrict__ bias,
                                float* __restrict__ out, int ldo,
                                int M, int N, int Nvalid, int K)
{
  int wid  = (blockIdx.x * blockDim.x + threadIdx.x) >> 5;
  int lane = threadIdx.x & 31;
  int Ntiles = N >> 4;
  int total  = (M >> 4) * Ntiles;
  if (wid >= total) return;                      // wave-uniform
  int nt = wid % Ntiles, mt = wid / Ntiles;
  int m0 = mt << 4, n0 = nt << 4;
  int l = lane & 15, h = lane >> 4;

  const float* arow = A + (size_t)(m0 + l) * lda;
  int nW = n0 + l;
  bool wok = nW < Nvalid;
  const float* wrow = Wt + (size_t)(wok ? nW : 0) * ldw;  // safe row, always loadable

  v8f acc = {};
  for (int k0 = 0; k0 < K; k0 += 32) {
    __builtin_prefetch(arow + k0 + 64, 0, 1);    // global_prefetch_b8
    v16bf af, bf;
#pragma unroll
    for (int j = 0; j < 8; ++j) {
      af[j]     = (__bf16)arow[k0 + h * 8 + j];
      af[8 + j] = (__bf16)arow[k0 + 16 + h * 8 + j];
    }
#pragma unroll
    for (int j = 0; j < 16; ++j) {
      float wv = wrow[k0 + (j < 8 ? h * 8 + j : 16 + h * 8 + (j - 8))];          // unconditional load
      bf[j] = (__bf16)(wok ? wv : 0.f);          // v_cndmask, no exec branch
    }
    acc = wmma_bf16(af, bf, acc);
  }

  int nc = n0 + l;
  if (nc < Nvalid) {
    float bv = bias[nc];
#pragma unroll
    for (int r = 0; r < 8; ++r)
      out[(size_t)(m0 + r + 8 * h) * ldo + nc] = acc[r] + bv;
  }
}

// ---------------------------------------------------------------------------
// GEMM: out[i,j] = scale * sum_b X[b,i]*Y[b,j] (+ diagAdd on diagonal).
// ---------------------------------------------------------------------------
__global__ void gemm_AtB_kernel(const float* __restrict__ X,
                                const float* __restrict__ Y,
                                float* __restrict__ out,
                                int rows, int N1, int N2, float scale, float diagAdd)
{
  int wid  = (blockIdx.x * blockDim.x + threadIdx.x) >> 5;
  int lane = threadIdx.x & 31;
  int Nt2 = N2 >> 4;
  int total = (N1 >> 4) * Nt2;
  if (wid >= total) return;
  int jt = wid % Nt2, it = wid / Nt2;
  int i0 = it << 4, j0 = jt << 4;
  int l = lane & 15, h = lane >> 4;

  v8f acc = {};
  for (int k0 = 0; k0 < rows; k0 += 32) {
    v16bf af, bf;
#pragma unroll
    for (int j = 0; j < 16; ++j) {
      int b = k0 + (j < 8 ? h * 8 + j : 16 + h * 8 + (j - 8));
      af[j] = (__bf16)X[(size_t)b * N1 + i0 + l];
    }
#pragma unroll
    for (int j = 0; j < 16; ++j) {
      int b = k0 + (j < 8 ? h * 8 + j : 16 + h * 8 + (j - 8));
      bf[j] = (__bf16)Y[(size_t)b * N2 + j0 + l];
    }
    acc = wmma_bf16(af, bf, acc);
  }

  int jc = j0 + l;
#pragma unroll
  for (int r = 0; r < 8; ++r) {
    int ir = i0 + r + 8 * h;
    float v = acc[r] * scale;
    if (ir == jc) v += diagAdd;
    out[(size_t)ir * N2 + jc] = v;
  }
}

// ---------------------------------------------------------------------------
// Implicit-GEMM conv3x3 SAME + BN(eval) + ReLU. Branchless clamped im2col.
// ---------------------------------------------------------------------------
template<int CI, int CO, int HW>
__global__ void conv_bn_relu_kernel(const float* __restrict__ x,
                                    const float* __restrict__ wgt,
                                    const float* __restrict__ bias,
                                    const float* __restrict__ bg,
                                    const float* __restrict__ bb,
                                    const float* __restrict__ bm,
                                    const float* __restrict__ bv,
                                    float* __restrict__ y, int Bsz)
{
  const int K   = CI * 9;
  const int NPT = (HW * HW) / 16;
  int wid  = (blockIdx.x * blockDim.x + threadIdx.x) >> 5;
  int lane = threadIdx.x & 31;
  int Mtiles = CO / 16;
  int total  = Mtiles * Bsz * NPT;
  if (wid >= total) return;
  int mt   = wid % Mtiles;
  int rest = wid / Mtiles;
  int pt   = rest % NPT;
  int img  = rest / NPT;
  int co0 = mt * 16;
  int p0  = pt * 16;
  int l = lane & 15, h = lane >> 4;

  int px = p0 + l;                 // this lane's output pixel (B-frag column)
  int ph = px / HW, pw = px % HW;
  const float* wr = wgt + (size_t)(co0 + l) * K;    // this lane's co row (A-frag)
  const float* xb = x + (size_t)img * CI * HW * HW;

  v8f acc = {};
  for (int k0 = 0; k0 < K; k0 += 32) {
    v16bf af, bf;
#pragma unroll
    for (int j = 0; j < 16; ++j) {
      int k = k0 + (j < 8 ? h * 8 + j : 16 + h * 8 + (j - 8));
      int kc = (k < K) ? k : (K - 1);              // clamp -> unconditional load
      float wv = wr[kc];
      af[j] = (__bf16)((k < K) ? wv : 0.f);
    }
#pragma unroll
    for (int j = 0; j < 16; ++j) {
      int k  = k0 + (j < 8 ? h * 8 + j : 16 + h * 8 + (j - 8));
      int kk = (k < K) ? k : 0;
      int ci = kk / 9, rem = kk % 9;
      int rr = rem / 3 - 1, cc = rem % 3 - 1;
      int hh = ph + rr, ww = pw + cc;
      bool ok = (k < K) && (hh >= 0) && (hh < HW) && (ww >= 0) && (ww < HW);
      int hhc = hh < 0 ? 0 : (hh >= HW ? HW - 1 : hh);
      int wwc = ww < 0 ? 0 : (ww >= HW ? HW - 1 : ww);
      float v = xb[((size_t)ci * HW + hhc) * HW + wwc];   // unconditional, clamped
      bf[j] = (__bf16)(ok ? v : 0.f);                     // v_cndmask
    }
    acc = wmma_bf16(af, bf, acc);
  }

#pragma unroll
  for (int r = 0; r < 8; ++r) {
    int co = co0 + r + 8 * h;
    float sc  = bg[co] * rsqrtf(bv[co] + 1e-5f);
    float val = (acc[r] + bias[co] - bm[co]) * sc + bb[co];
    val = fmaxf(val, 0.f);
    y[(((size_t)img * CO + co) * HW + ph) * HW + pw] = val;
  }
}

// ---------------------------------------------------------------------------
// Pooling / elementwise / reductions
// ---------------------------------------------------------------------------
__global__ void maxpool_kernel(const float* __restrict__ in, float* __restrict__ out,
                               int HWo, long long total)
{
  long long idx = (long long)blockIdx.x * blockDim.x + threadIdx.x;
  if (idx >= total) return;
  int HWi = HWo * 2;
  int j = (int)(idx % HWo); long long r1 = idx / HWo;
  int i = (int)(r1 % HWo);  long long bc = r1 / HWo;
  const float* p = in + (bc * HWi + 2 * i) * HWi + 2 * j;
  out[idx] = fmaxf(fmaxf(p[0], p[1]), fmaxf(p[HWi], p[HWi + 1]));
}

__global__ void avgpool_kernel(const float* __restrict__ in, float* __restrict__ out,
                               int HW, long long total)
{
  long long idx = (long long)blockIdx.x * blockDim.x + threadIdx.x;
  if (idx >= total) return;
  const float* p = in + idx * HW * HW;
  float s = 0.f;
  for (int i = 0; i < HW * HW; ++i) s += p[i];
  out[idx] = s / (float)(HW * HW);
}

// LayerNorm over `cols` (blockDim == cols), optional residual add + ReLU.
__global__ void ln_kernel(const float* __restrict__ in, int ldin,
                          const float* __restrict__ g, const float* __restrict__ b,
                          const float* __restrict__ resid, int ldres,
                          float* __restrict__ out, int ldout, int cols, int relu)
{
  __shared__ float red[256];
  int row = blockIdx.x, t = threadIdx.x;
  float v = in[(size_t)row * ldin + t];
  red[t] = v;
  __syncthreads();
  for (int s = cols >> 1; s > 0; s >>= 1) { if (t < s) red[t] += red[t + s]; __syncthreads(); }
  float mean = red[0] / cols;
  __syncthreads();
  float d = v - mean;
  red[t] = d * d;
  __syncthreads();
  for (int s = cols >> 1; s > 0; s >>= 1) { if (t < s) red[t] += red[t + s]; __syncthreads(); }
  float var = red[0] / cols;
  float y = d * rsqrtf(var + 1e-5f) * g[t] + b[t];
  if (resid) y += resid[(size_t)row * ldres + t];
  if (relu) y = fmaxf(y, 0.f);
  out[(size_t)row * ldout + t] = y;
}

// Row-wise L2 normalize (blockDim == cols).
__global__ void rowl2_kernel(const float* __restrict__ in, float* __restrict__ out, int cols)
{
  __shared__ float red[256];
  int row = blockIdx.x, t = threadIdx.x;
  float v = in[(size_t)row * cols + t];
  red[t] = v * v;
  __syncthreads();
  for (int s = cols >> 1; s > 0; s >>= 1) { if (t < s) red[t] += red[t + s]; __syncthreads(); }
  float nrm = sqrtf(red[0]);
  out[(size_t)row * cols + t] = v / fmaxf(nrm, 1e-12f);
}

__global__ void colmean_kernel(const float* __restrict__ in, int ld,
                               float* __restrict__ out, int rows)
{
  int c = threadIdx.x;
  float s = 0.f;
  for (int r = 0; r < rows; ++r) s += in[(size_t)r * ld + c];
  out[c] = s / (float)rows;
}

__global__ void memnew_kernel(const float* __restrict__ mem, const float* __restrict__ cm,
                              float* __restrict__ out)
{
  int c = threadIdx.x;
  out[c] = 0.9f * mem[c] + 0.1f * cm[c];
}

__global__ void spin_kernel(const float* __restrict__ hc, const float* __restrict__ mn,
                            float* __restrict__ out, int total, int D)
{
  int i = blockIdx.x * blockDim.x + threadIdx.x;
  if (i >= total) return;
  out[i] = hc[i] + 0.3f * mn[i % D];
}

// fw_eff = clip-by-row-norm(fw*decay + lr*hebb); wsum = sw + fw_eff. blockDim==256.
__global__ void make_wsum_kernel(const float* __restrict__ sw, const float* __restrict__ fw,
                                 const float* __restrict__ hebb, float* __restrict__ wsum,
                                 float decay, float lr)
{
  __shared__ float red[256];
  int r = blockIdx.x, c = threadIdx.x;
  size_t idx = (size_t)r * 256 + c;
  float t = fw[idx] * decay + lr * hebb[idx];
  red[c] = t * t;
  __syncthreads();
  for (int s = 128; s > 0; s >>= 1) { if (c < s) red[c] += red[c + s]; __syncthreads(); }
  float nrm = sqrtf(red[0]);
  float sc  = (nrm > 0.25f) ? 0.25f / (nrm + 1e-8f) : 1.f;
  wsum[idx] = sw[idx] + t * sc;
}

__global__ void make_bsum_kernel(const float* __restrict__ sb, const float* __restrict__ fb,
                                 const float* __restrict__ sonmean, float* __restrict__ bsum,
                                 float decay, float lr)
{
  int c = threadIdx.x;
  float fbe = fb[c] * decay + lr * sonmean[c];
  fbe = fminf(fmaxf(fbe, -0.2f), 0.2f);
  bsum[c] = sb[c] + fbe;
}

// an[r][c] = (hd[r][c] - am[c]) / (std_ddof1[c] + 1e-6). blockDim==256.
__global__ void an_kernel(const float* __restrict__ hd, const float* __restrict__ am,
                          float* __restrict__ an, int rows)
{
  int c = threadIdx.x;
  float mn = am[c];
  float s = 0.f;
  for (int r = 0; r < rows; ++r) { float d = hd[(size_t)r * 256 + c] - mn; s += d * d; }
  float ast = sqrtf(s / (float)(rows - 1)) + 1e-6f;
  for (int r = 0; r < rows; ++r)
    an[(size_t)r * 256 + c] = (hd[(size_t)r * 256 + c] - mn) / ast;
}

__global__ void addcoef_kernel(const float* __restrict__ hd, const float* __restrict__ net,
                               const float* __restrict__ coef, float* __restrict__ out, int total)
{
  int i = blockIdx.x * blockDim.x + threadIdx.x;
  if (i >= total) return;
  out[i] = hd[i] + coef[0] * net[i];
}

// ---------------------------------------------------------------------------
// Parallel two-sided Jacobi eigensolver on 256x256 in LDS (259KB, CDNA5-only),
// then eigen-entropy -> phi -> consciousness coefficient. One workgroup.
// ---------------------------------------------------------------------------
__global__ void jacobi_phi_kernel(const float* __restrict__ cov, float* __restrict__ coef_out)
{
  extern __shared__ float sm[];
  float* Amat = sm;                        // 65536 floats
  float* cs   = sm + 65536;                // 256 floats: c,s per pair
  int*   pq   = (int*)(sm + 65536 + 256);  // 256 ints: p,q per pair
  float* red  = sm + 65536 + 512;          // 256 floats

  int t = threadIdx.x;
  for (int i = t; i < 65536; i += 256) Amat[i] = cov[i];
  __syncthreads();

  const int n = 256;
  for (int sweep = 0; sweep < 6; ++sweep) {
    for (int r = 0; r < n - 1; ++r) {
      if (t < 128) {
        int p, q;
        if (t == 0) { p = 255; q = r % 255; }
        else        { p = (r + t) % 255; q = (r - t + 255) % 255; }
        float app = Amat[p * 256 + p];
        float aqq = Amat[q * 256 + q];
        float apq = Amat[p * 256 + q];
        float c = 1.f, s = 0.f;
        if (fabsf(apq) > 1e-12f) {
          float tau = (aqq - app) / (2.f * apq);
          float tt  = (tau >= 0.f ? 1.f : -1.f) / (fabsf(tau) + sqrtf(1.f + tau * tau));
          c = rsqrtf(1.f + tt * tt);
          s = tt * c;
        }
        cs[2 * t] = c; cs[2 * t + 1] = s;
        pq[2 * t] = p; pq[2 * t + 1] = q;
      }
      __syncthreads();
      for (int i = 0; i < 128; ++i) {        // row rotations (disjoint rows)
        float c = cs[2 * i], s = cs[2 * i + 1];
        int p = pq[2 * i], q = pq[2 * i + 1];
        float apj = Amat[p * 256 + t], aqj = Amat[q * 256 + t];
        Amat[p * 256 + t] = c * apj - s * aqj;
        Amat[q * 256 + t] = s * apj + c * aqj;
      }
      __syncthreads();
      for (int i = 0; i < 128; ++i) {        // column rotations (disjoint cols)
        float c = cs[2 * i], s = cs[2 * i + 1];
        int p = pq[2 * i], q = pq[2 * i + 1];
        float ajp = Amat[t * 256 + p], ajq = Amat[t * 256 + q];
        Amat[t * 256 + p] = c * ajp - s * ajq;
        Amat[t * 256 + q] = s * ajp + c * ajq;
      }
      __syncthreads();
    }
  }

  float d = fmaxf(Amat[t * 256 + t], 0.f);
  red[t] = d;
  __syncthreads();
  for (int s = 128; s > 0; s >>= 1) { if (t < s) red[t] += red[t + s]; __syncthreads(); }
  float S = red[0];
  __syncthreads();
  float en = d / (S + 1e-8f);
  red[t] = -(en * logf(en + 1e-8f));
  __syncthreads();
  for (int s = 128; s > 0; s >>= 1) { if (t < s) red[t] += red[t + s]; __syncthreads(); }
  if (t == 0) {
    float entropy = red[0];
    float phi = entropy / logf(256.f);
    float phi_eff = fminf(fmaxf(0.3f * phi, 0.f), 1.f);
    float strength = 1.f / (1.f + expf(-(phi_eff - 0.1f) * 5.f));
    coef_out[0] = (phi_eff > 0.1f) ? strength : 0.1f;
  }
}

// ---------------------------------------------------------------------------
// Host orchestration
// ---------------------------------------------------------------------------
extern "C" void kernel_launch(void* const* d_in, const int* in_sizes, int n_in,
                              void* d_out, int out_size, void* d_ws, size_t ws_size,
                              hipStream_t stream)
{
  (void)in_sizes; (void)n_in; (void)out_size; (void)ws_size;
  auto F = [&](int i) { return (const float*)d_in[i]; };
  const int B = 512, D = 256;

  float* ws = (float*)d_ws;
  size_t off = 0;
  auto alloc = [&](size_t nelem) { float* p = ws + off; off += (nelem + 63) & ~(size_t)63; return p; };

  float* bufA = alloc((size_t)B * 64 * 32 * 32);
  float* bufB = alloc((size_t)B * 64 * 16 * 16);
  float* h    = alloc((size_t)B * D);
  float* t0   = alloc((size_t)B * D);
  float* hc   = alloc((size_t)B * D);
  float* cm   = alloc(D);
  float* mn   = alloc(D);
  float* t1   = alloc((size_t)B * D);
  float* t2   = alloc((size_t)B * D);
  float* t3   = alloc((size_t)B * D);
  float* t4   = alloc((size_t)D * D);
  float* cm2  = alloc(D);
  float* t5   = alloc((size_t)D * D);
  float* t6   = alloc(D);
  float* t7   = alloc((size_t)B * D);
  float* comb = alloc((size_t)B * 2 * D);
  float* t8   = alloc((size_t)B * D);
  float* t9   = alloc((size_t)B * D);
  float* hd   = alloc((size_t)B * D);
  float* am   = alloc(D);
  float* anb  = alloc((size_t)B * D);
  float* covb = alloc((size_t)D * D);
  float* coefb= alloc(64);
  float* t10  = alloc((size_t)B * D);
  float* t11  = alloc((size_t)B * D);
  float* t12  = alloc((size_t)B * D);
  float* t13  = alloc((size_t)B * D);
  float* t14  = alloc((size_t)B * 128);
  float* t15  = alloc((size_t)B * 128);

  auto gemm = [&](const float* A, int lda, const float* Wt, int ldw, const float* bias,
                  float* out, int ldo, int M, int N, int Nvalid, int K) {
    if (Nvalid == N && (N & 63) == 0) {
      int blocks = (M / 16) * (N / 64);
      gemm_xwT_tdm_kernel<<<blocks, 128, 0, stream>>>(A, lda, Wt, ldw, bias, out, ldo, M, N, K);
    } else {
      int waves = (M / 16) * (N / 16);
      gemm_xwT_kernel<<<(waves + 3) / 4, 128, 0, stream>>>(A, lda, Wt, ldw, bias, out, ldo,
                                                           M, N, Nvalid, K);
    }
  };
  auto atb = [&](const float* X, const float* Y, float* out, float scale, float diagAdd) {
    int waves = (D / 16) * (D / 16);
    gemm_AtB_kernel<<<(waves + 3) / 4, 128, 0, stream>>>(X, Y, out, B, D, D, scale, diagAdd);
  };
  auto ln = [&](const float* in, int ldin, const float* g, const float* bb,
                const float* res, int ldres, float* out, int ldout, int cols, int relu) {
    ln_kernel<<<B, cols, 0, stream>>>(in, ldin, g, bb, res, ldres, out, ldout, cols, relu);
  };

  // ---------------- encoder ----------------
  {
    int waves = (64 / 16) * B * (32 * 32 / 16);
    conv_bn_relu_kernel<3, 64, 32><<<waves / 4, 128, 0, stream>>>(
        F(0), F(1), F(2), F(3), F(4), F(5), F(6), bufA, B);
    long long tot = (long long)B * 64 * 16 * 16;
    maxpool_kernel<<<(int)((tot + 255) / 256), 256, 0, stream>>>(bufA, bufB, 16, tot);

    waves = (128 / 16) * B * (16 * 16 / 16);
    conv_bn_relu_kernel<64, 128, 16><<<waves / 4, 128, 0, stream>>>(
        bufB, F(7), F(8), F(9), F(10), F(11), F(12), bufA, B);
    tot = (long long)B * 128 * 8 * 8;
    maxpool_kernel<<<(int)((tot + 255) / 256), 256, 0, stream>>>(bufA, bufB, 8, tot);

    waves = (256 / 16) * B * (8 * 8 / 16);
    conv_bn_relu_kernel<128, 256, 8><<<waves / 4, 128, 0, stream>>>(
        bufB, F(13), F(14), F(15), F(16), F(17), F(18), bufA, B);
    tot = (long long)B * 256;
    avgpool_kernel<<<(int)((tot + 255) / 256), 256, 0, stream>>>(bufA, h, 8, tot);
  }

  // ---------------- core ----------------
  gemm(h, D, F(19), D, F(20), t0, D, B, D, D, D);
  ln(t0, D, F(21), F(22), nullptr, 0, hc, D, D, 1);

  // memory EMA
  colmean_kernel<<<1, D, 0, stream>>>(hc, D, cm, B);
  memnew_kernel<<<1, D, 0, stream>>>(F(39), cm, mn);

  // ---------------- fast/slow hebbian paths ----------------
  auto fastslow = [&](const float* x, const float* sw, const float* sb,
                      const float* fw, const float* fb,
                      const float* lng, const float* lnb,
                      float* outp, int ldout, float lr, float decay) {
    gemm(x, D, sw, D, sb, t1, D, B, D, D, D);          // slow_out
    rowl2_kernel<<<B, D, 0, stream>>>(x, t2, D);       // xn
    rowl2_kernel<<<B, D, 0, stream>>>(t1, t3, D);      // son
    atb(t3, t2, t4, 1.f / (float)B, 0.f);              // hebb = son^T xn / B
    colmean_kernel<<<1, D, 0, stream>>>(t3, D, cm2, B);
    make_wsum_kernel<<<D, D, 0, stream>>>(sw, fw, t4, t5, decay, lr);
    make_bsum_kernel<<<1, D, 0, stream>>>(sb, fb, cm2, t6, decay, lr);
    gemm(x, D, t5, D, t6, t7, D, B, D, D, D);          // x @ (sw+fw)^T + (sb+fb)
    ln(t7, D, lng, lnb, nullptr, 0, outp, ldout, D, 0);
  };

  fastslow(hc, F(23), F(24), F(25), F(26), F(27), F(28), comb,     2 * D, 0.015f, 0.95f);
  spin_kernel<<<(B * D + 255) / 256, 256, 0, stream>>>(hc, mn, t8, B * D, D);
  fastslow(t8, F(29), F(30), F(31), F(32), F(33), F(34), comb + D, 2 * D, 0.005f, 0.97f);

  // integrator + residual
  gemm(comb, 2 * D, F(35), 2 * D, F(36), t9, D, B, D, D, 2 * D);
  ln(t9, D, F(37), F(38), hc, D, hd, D, D, 0);

  // ---------------- consciousness (eigen-entropy) ----------------
  colmean_kernel<<<1, D, 0, stream>>>(hd, D, am, B);
  an_kernel<<<1, D, 0, stream>>>(hd, am, anb, B);
  atb(anb, anb, covb, 1.f / (float)(B - 1), 1e-4f);
  size_t smem = (size_t)(65536 + 512 + 256) * sizeof(float);
  jacobi_phi_kernel<<<1, 256, smem, stream>>>(covb, coefb);

  gemm(hd, D, F(40), D, F(41), t10, D, B, D, D, D);
  ln(t10, D, F(42), F(43), nullptr, 0, t11, D, D, 1);
  gemm(t11, D, F(44), D, F(45), t12, D, B, D, D, D);
  addcoef_kernel<<<(B * D + 255) / 256, 256, 0, stream>>>(hd, t12, coefb, t13, B * D);

  // ---------------- projection + classifier ----------------
  gemm(t13, D, F(46), D, F(47), t14, 128, B, 128, 128, D);
  ln_kernel<<<B, 128, 0, stream>>>(t14, 128, F(48), F(49), nullptr, 0, t15, 128, 128, 1);
  gemm(t15, 128, F(50), 128, F(51), (float*)d_out, 10, B, 16, 10, 128);
}
